// MoEClassifier_86380382257486
// MI455X (gfx1250) — matrix operand order repaired and checked by reading, compiled, not run on hardware
//
#include <hip/hip_runtime.h>

typedef __attribute__((ext_vector_type(16))) __bf16 v16bf;
typedef __attribute__((ext_vector_type(8)))  float  v8f;
typedef __attribute__((ext_vector_type(4)))  unsigned int u32x4;

constexpr int B_   = 8192;
constexpr int DIN  = 1024;
constexpr int HID  = 4096;
constexpr int C_   = 1000;
constexpr int CP   = 1024;   // padded C
constexpr int E_   = 8;
constexpr int K_   = 2;
constexpr int BK   = B_ * K_;

// GEMM tiling
constexpr int BM    = 128;
constexpr int BN    = 256;
constexpr int KSTEP = 32;
constexpr int KP    = 48;    // LDS k-stride in halves (96B rows: 16B aligned, staggers banks)

__device__ __forceinline__ unsigned short f2bf(float f) {
  unsigned int u = __float_as_uint(f);
  u += 0x7fffu + ((u >> 16) & 1u);
  return (unsigned short)(u >> 16);
}

// ---- CDNA5 async global->LDS copy (ASYNCcnt path, CDNA5 ISA) ----
__device__ __forceinline__ void async_cp16(unsigned lds_off, const void* gaddr) {
  asm volatile("global_load_async_to_lds_b128 %0, %1, off"
               :: "v"(lds_off), "v"(gaddr) : "memory");
}
#if __has_builtin(__builtin_amdgcn_s_wait_asynccnt)
#define WAIT_ASYNC() __builtin_amdgcn_s_wait_asynccnt(0)
#else
#define WAIT_ASYNC() asm volatile("s_wait_asynccnt 0" ::: "memory")
#endif

// ---------------- small utility kernels ----------------

__global__ void zero_small_kernel(int* counts, float* prob_sums) {
  int t = threadIdx.x;
  if (t < E_) { counts[t] = 0; prob_sums[t] = 0.f; }
}

__global__ void cvt_bf16_kernel(const float* __restrict__ src,
                                unsigned short* __restrict__ dst, int n4) {
  int stride = gridDim.x * blockDim.x;
  for (int i = blockIdx.x * blockDim.x + threadIdx.x; i < n4; i += stride) {
    float4 v = ((const float4*)src)[i];
    ushort4 o;
    o.x = f2bf(v.x); o.y = f2bf(v.y); o.z = f2bf(v.z); o.w = f2bf(v.w);
    ((ushort4*)dst)[i] = o;
  }
}

// f32 [E][Kdim][Nsrc] -> bf16 [E][Ndst][Kdim]  (transpose + pad cols>=Nsrc with 0)
__global__ __launch_bounds__(256) void cvt_transpose_kernel(
    const float* __restrict__ src, unsigned short* __restrict__ dst,
    int Kdim, int Nsrc, int Ndst) {
  __shared__ float tile[64][65];
  const int ktiles = Kdim >> 6, ntiles = Ndst >> 6;
  int bid = blockIdx.x;
  int e = bid / (ktiles * ntiles);
  int r = bid % (ktiles * ntiles);
  int k0 = (r / ntiles) << 6;
  int n0 = (r % ntiles) << 6;
  const float* se = src + (long long)e * Kdim * Nsrc;
#pragma unroll
  for (int j = 0; j < 4; ++j) {
    int idx = threadIdx.x + j * 256;   // 0..1023 -> one float4 each
    int kk = idx >> 4;
    int c4 = (idx & 15) << 2;
    int n = n0 + c4;
    float4 v;
    if (n + 3 < Nsrc) {
      v = *(const float4*)(se + (long long)(k0 + kk) * Nsrc + n);
    } else {
      const float* rowp = se + (long long)(k0 + kk) * Nsrc;
      v.x = (n + 0 < Nsrc) ? rowp[n + 0] : 0.f;
      v.y = (n + 1 < Nsrc) ? rowp[n + 1] : 0.f;
      v.z = (n + 2 < Nsrc) ? rowp[n + 2] : 0.f;
      v.w = (n + 3 < Nsrc) ? rowp[n + 3] : 0.f;
    }
    tile[kk][c4 + 0] = v.x; tile[kk][c4 + 1] = v.y;
    tile[kk][c4 + 2] = v.z; tile[kk][c4 + 3] = v.w;
  }
  __syncthreads();
  unsigned short* de = dst + (long long)e * Ndst * Kdim;
#pragma unroll
  for (int j = 0; j < 2; ++j) {
    int idx = threadIdx.x + j * 256;   // 0..511 -> one 8-half chunk each
    int nn = idx >> 3;
    int kc = (idx & 7) << 3;
    ushort4 o0, o1;
    o0.x = f2bf(tile[kc + 0][nn]); o0.y = f2bf(tile[kc + 1][nn]);
    o0.z = f2bf(tile[kc + 2][nn]); o0.w = f2bf(tile[kc + 3][nn]);
    o1.x = f2bf(tile[kc + 4][nn]); o1.y = f2bf(tile[kc + 5][nn]);
    o1.z = f2bf(tile[kc + 6][nn]); o1.w = f2bf(tile[kc + 7][nn]);
    ushort4* d = (ushort4*)(de + (long long)(n0 + nn) * Kdim + k0 + kc);
    d[0] = o0; d[1] = o1;
  }
}

// ---------------- gating: logits, softmax, top-2, routing ----------------

__global__ void gating_kernel(const float* __restrict__ x, const float* __restrict__ Wg,
                              const float* __restrict__ bg,
                              int* __restrict__ counts, int* __restrict__ token_list,
                              int* __restrict__ route_e, int* __restrict__ route_pos,
                              float* __restrict__ route_w, float* __restrict__ prob_sums) {
  int wid = threadIdx.x >> 5;
  int lane = threadIdx.x & 31;
  int b = blockIdx.x * 8 + wid;
  if (b >= B_) return;

  float acc[E_];
#pragma unroll
  for (int e = 0; e < E_; ++e) acc[e] = 0.f;
  const float* xr = x + (long long)b * DIN;
  for (int i = lane; i < DIN; i += 32) {
    float xv = xr[i];
    const float* wr = Wg + i * E_;
#pragma unroll
    for (int e = 0; e < E_; ++e) acc[e] += xv * wr[e];
  }
#pragma unroll
  for (int e = 0; e < E_; ++e) {
#pragma unroll
    for (int off = 16; off; off >>= 1) acc[e] += __shfl_xor(acc[e], off, 32);
  }
  if (lane == 0) {
    float p[E_];
    float m = -3.4e38f;
#pragma unroll
    for (int e = 0; e < E_; ++e) { p[e] = acc[e] + bg[e]; m = p[e] > m ? p[e] : m; }
    float s = 0.f;
#pragma unroll
    for (int e = 0; e < E_; ++e) { p[e] = __expf(p[e] - m); s += p[e]; }
    float inv = 1.f / s;
#pragma unroll
    for (int e = 0; e < E_; ++e) { p[e] *= inv; atomicAdd(&prob_sums[e], p[e]); }
    int i1 = 0;
#pragma unroll
    for (int e = 1; e < E_; ++e) if (p[e] > p[i1]) i1 = e;
    int i2 = (i1 == 0) ? 1 : 0;
#pragma unroll
    for (int e = 0; e < E_; ++e) if (e != i1 && p[e] > p[i2]) i2 = e;
    float denom = p[i1] + p[i2];
    float w0 = p[i1] / denom, w1 = p[i2] / denom;

    int pos1 = atomicAdd(&counts[i1], 1);
    token_list[i1 * B_ + pos1] = b;
    route_e[b * 2] = i1; route_pos[b * 2] = pos1; route_w[b * 2] = w0;
    int pos2 = atomicAdd(&counts[i2], 1);
    token_list[i2 * B_ + pos2] = b;
    route_e[b * 2 + 1] = i2; route_pos[b * 2 + 1] = pos2; route_w[b * 2 + 1] = w1;
  }
}

// ---------------- tiny scan: offsets, tile offsets, lb_loss ----------------

__global__ void scan_kernel(const int* __restrict__ counts, int* __restrict__ h_off,
                            int* __restrict__ tiles_off, const float* __restrict__ prob_sums,
                            float* __restrict__ out_lb) {
  if (threadIdx.x == 0 && blockIdx.x == 0) {
    int ro = 0, to = 0;
    for (int e = 0; e < E_; ++e) {
      h_off[e] = ro; tiles_off[e] = to;
      ro += counts[e];
      to += (counts[e] + BM - 1) / BM;
    }
    h_off[E_] = ro; tiles_off[E_] = to;
    float lb = 0.f;
    for (int e = 0; e < E_; ++e) { float mm = prob_sums[e] / (float)B_; lb += mm * mm; }
    out_lb[0] = lb * (float)E_;
  }
}

// ---------------- routed expert GEMM (bf16 WMMA, async-LDS double buffer) ----------------
// A: [rows, KDIM] bf16 (GATHER -> rows through token_list)
// B: [E][NDIM][KDIM] bf16 (pre-transposed, n-major -> contiguous 64B tile rows)

template <bool GATHER, bool RELU_BF16_OUT>
__global__ __launch_bounds__(256) void moe_gemm_kernel(
    const unsigned short* __restrict__ Amat,
    const unsigned short* __restrict__ Bmat,
    const float* __restrict__ bias,            // [E, bias_cols]
    void* __restrict__ Out,                    // bf16 or f32, [.., NDIM]
    const int* __restrict__ counts, const int* __restrict__ h_off,
    const int* __restrict__ tiles_off, const int* __restrict__ token_list,
    int KDIM, int NDIM, int bias_cols) {
  __shared__ unsigned short As[2][BM * KP];   // [m][k]
  __shared__ unsigned short Bs[2][BN * KP];   // [n][k]

  const int CT = NDIM / BN;
  const int tid = threadIdx.x;
  const int lane = tid & 31;
  const int wid = tid >> 5;
  const int wm = wid >> 2;     // 0..1 -> 64-row slab
  const int wn = wid & 3;      // 0..3 -> 64-col slab
  const int lhalf = lane >> 4;
  const int lmod = lane & 15;
  const int total_tiles = tiles_off[E_] * CT;

  for (int t = blockIdx.x; t < total_tiles; t += gridDim.x) {
    int rt = t / CT;
    int ct = t % CT;
    int e = 0;
#pragma unroll
    for (int i = 1; i < E_; ++i) if (tiles_off[i] <= rt) e = i;
    const int row_start = (rt - tiles_off[e]) * BM;
    int nrows = counts[e] - row_start; if (nrows > BM) nrows = BM;
    const int nbase = ct * BN;
    const int out_row_base = h_off[e] + row_start;
    const unsigned short* Bexp = Bmat + (long long)e * KDIM * NDIM;

    // A-load assignment: row = tid/2, two 16B chunks; B: row = tid, four 16B chunks
    const int arow = tid >> 1;
    const int achunk0 = (tid & 1) * 2;
    long long aRowBase;
    {
      int r = arow < nrows ? arow : 0;
      int rglob = GATHER ? token_list[e * B_ + row_start + r] : (out_row_base + r);
      aRowBase = (long long)rglob * KDIM;
    }
    const unsigned short* bRowBase = Bexp + (long long)(nbase + tid) * KDIM;

    auto issue_tile = [&](int kbase, int bsel) {
      const unsigned short* ga = Amat + aRowBase + kbase + achunk0 * 8;
      unsigned la = (unsigned)(size_t)&As[bsel][arow * KP + achunk0 * 8];
      async_cp16(la, ga);
      async_cp16(la + 16, ga + 8);
      const unsigned short* gb = bRowBase + kbase;
      unsigned lb = (unsigned)(size_t)&Bs[bsel][tid * KP];
      async_cp16(lb,      gb);
      async_cp16(lb + 16, gb + 8);
      async_cp16(lb + 32, gb + 16);
      async_cp16(lb + 48, gb + 24);
    };

    // accumulators initialized with bias (broadcast over rows)
    v8f acc[4][4];
#pragma unroll
    for (int ni = 0; ni < 4; ++ni) {
      int col = nbase + wn * 64 + ni * 16 + lmod;
      float bv = (col < bias_cols) ? bias[(long long)e * bias_cols + col] : 0.f;
#pragma unroll
      for (int mi = 0; mi < 4; ++mi)
#pragma unroll
        for (int r = 0; r < 8; ++r) acc[mi][ni][r] = bv;
    }

    const int KT = KDIM / KSTEP;
    int buf = 0;
    issue_tile(0, 0);
    for (int kt = 0; kt < KT; ++kt) {
      WAIT_ASYNC();          // this wave's copies into As[buf]/Bs[buf] complete
      __syncthreads();       // all waves' copies visible; prev buffer's readers done
      if (kt + 1 < KT) issue_tile((kt + 1) * KSTEP, buf ^ 1);

      union FB { v16bf v; u32x4 u[2]; };
      FB af[4], bfr[4];
#pragma unroll
      for (int mi = 0; mi < 4; ++mi) {
        int m = wm * 64 + mi * 16 + lmod;
        af[mi].u[0] = *(const u32x4*)(&As[buf][m * KP + lhalf * 8]);
        af[mi].u[1] = *(const u32x4*)(&As[buf][m * KP + 16 + lhalf * 8]);
      }
#pragma unroll
      for (int ni = 0; ni < 4; ++ni) {
        int n = wn * 64 + ni * 16 + lmod;
        bfr[ni].u[0] = *(const u32x4*)(&Bs[buf][n * KP + lhalf * 8]);
        bfr[ni].u[1] = *(const u32x4*)(&Bs[buf][n * KP + 16 + lhalf * 8]);
      }
#pragma unroll
      for (int ni = 0; ni < 4; ++ni) {
#pragma unroll
        for (int mi = 0; mi < 4; ++mi) {
          acc[mi][ni] = __builtin_amdgcn_wmma_f32_16x16x32_bf16(
              false, af[mi].v, false, bfr[ni].v, (short)0, acc[mi][ni], false, false);
        }
      }
      buf ^= 1;
    }

    // epilogue
#pragma unroll
    for (int mi = 0; mi < 4; ++mi) {
#pragma unroll
      for (int r = 0; r < 8; ++r) {
        int mlocal = wm * 64 + mi * 16 + r + lhalf * 8;
        if (mlocal < nrows) {
          long long orow = (long long)(out_row_base + mlocal) * NDIM;
#pragma unroll
          for (int ni = 0; ni < 4; ++ni) {
            int col = nbase + wn * 64 + ni * 16 + lmod;
            float v = acc[mi][ni][r];
            if (RELU_BF16_OUT) {
              v = v > 0.f ? v : 0.f;
              ((unsigned short*)Out)[orow + col] = f2bf(v);
            } else {
              ((float*)Out)[orow + col] = v;
            }
          }
        }
      }
    }
  }
}

// ---------------- final combine: out[b,:] = w0*y[r0,:] + w1*y[r1,:] ----------------

__global__ void combine_kernel(const float* __restrict__ y, const int* __restrict__ route_e,
                               const int* __restrict__ route_pos, const float* __restrict__ route_w,
                               const int* __restrict__ h_off, float* __restrict__ out) {
  int b = blockIdx.x;
  int e0 = route_e[b * 2], e1 = route_e[b * 2 + 1];
  long long r0 = h_off[e0] + route_pos[b * 2];
  long long r1 = h_off[e1] + route_pos[b * 2 + 1];
  float w0 = route_w[b * 2], w1 = route_w[b * 2 + 1];
  const float* y0 = y + r0 * CP;
  const float* y1 = y + r1 * CP;
  float* o = out + (long long)b * C_;
  for (int c = threadIdx.x; c < C_; c += blockDim.x)
    o[c] = w0 * y0[c] + w1 * y1[c];
}

// ---------------- launch ----------------

extern "C" void kernel_launch(void* const* d_in, const int* in_sizes, int n_in,
                              void* d_out, int out_size, void* d_ws, size_t ws_size,
                              hipStream_t stream) {
  const float* x  = (const float*)d_in[0];
  const float* Wg = (const float*)d_in[1];
  const float* bg = (const float*)d_in[2];
  const float* W1 = (const float*)d_in[3];
  const float* b1 = (const float*)d_in[4];
  const float* W2 = (const float*)d_in[5];
  const float* b2 = (const float*)d_in[6];
  float* out = (float*)d_out;
  (void)in_sizes; (void)n_in; (void)out_size; (void)ws_size;

  char* ws = (char*)d_ws;
  auto alloc = [&](size_t bytes) {
    char* p = ws;
    ws += (bytes + 255) & ~(size_t)255;
    return p;
  };
  unsigned short* xb   = (unsigned short*)alloc((size_t)B_ * DIN * 2);
  unsigned short* W1t  = (unsigned short*)alloc((size_t)E_ * DIN * HID * 2);  // [E][HID][DIN]
  unsigned short* W2t  = (unsigned short*)alloc((size_t)E_ * HID * CP * 2);   // [E][CP][HID]
  unsigned short* hbuf = (unsigned short*)alloc((size_t)BK * HID * 2);
  float*          ybuf = (float*)alloc((size_t)BK * CP * 4);
  int*   token_list = (int*)alloc((size_t)E_ * B_ * 4);
  int*   counts     = (int*)alloc(64);
  int*   h_off      = (int*)alloc(64);
  int*   tiles_off  = (int*)alloc(64);
  int*   route_e    = (int*)alloc((size_t)BK * 4);
  int*   route_pos  = (int*)alloc((size_t)BK * 4);
  float* route_w    = (float*)alloc((size_t)BK * 4);
  float* prob_sums  = (float*)alloc(64);

  zero_small_kernel<<<1, 32, 0, stream>>>(counts, prob_sums);
  cvt_bf16_kernel<<<4096, 256, 0, stream>>>(x, xb, B_ * DIN / 4);
  cvt_transpose_kernel<<<8192, 256, 0, stream>>>(W1, W1t, DIN, HID, HID);
  cvt_transpose_kernel<<<8192, 256, 0, stream>>>(W2, W2t, HID, C_, CP);
  gating_kernel<<<B_ / 8, 256, 0, stream>>>(x, Wg, bg, counts, token_list,
                                            route_e, route_pos, route_w, prob_sums);
  scan_kernel<<<1, 1, 0, stream>>>(counts, h_off, tiles_off, prob_sums,
                                   out + (size_t)B_ * C_);
  moe_gemm_kernel<true, true><<<2048, 256, 0, stream>>>(
      xb, W1t, b1, hbuf, counts, h_off, tiles_off, token_list, DIN, HID, HID);
  moe_gemm_kernel<false, false><<<1024, 256, 0, stream>>>(
      hbuf, W2t, b2, ybuf, counts, h_off, tiles_off, token_list, HID, CP, C_);
  combine_kernel<<<B_, 256, 0, stream>>>(ybuf, route_e, route_pos, route_w, h_off, out);
}
